// TernaryLinear_31868657336979
// MI455X (gfx1250) — matrix operand order repaired and checked by reading, not compiled, not run on hardware
//
#include <hip/hip_runtime.h>

// ---------------------------------------------------------------------------
// BitNet ternary linear for gfx1250 (MI455X):
//   alpha = mean(|W|)+eps ; Wt = clip(rint(W/alpha),-1,1) (exact in f16)
//   out   = alpha * (Xf16 @ Wt_f16^T)  via v_wmma_f32_16x16x32_f16
// ---------------------------------------------------------------------------

typedef _Float16 v4h  __attribute__((ext_vector_type(4)));
typedef _Float16 v8h  __attribute__((ext_vector_type(8)));
typedef _Float16 v16h __attribute__((ext_vector_type(16)));
typedef float    v4f  __attribute__((ext_vector_type(4)));
typedef float    v8f  __attribute__((ext_vector_type(8)));

#define IN_F   2048
#define OUT_F  4096
#define NTOK   8192
#define WELEMS (OUT_F * IN_F)   // 8,388,608
#define XELEMS (NTOK  * IN_F)   // 16,777,216

// ---------------- 1) deterministic two-pass mean(|W|) ----------------------

__global__ __launch_bounds__(256)
void bitnet_abs_partial(const float* __restrict__ w, float* __restrict__ partials) {
    __shared__ float sm[256];
    const int tid = threadIdx.x;
    float s = 0.f;
    for (size_t i = (size_t)blockIdx.x * 256 + tid; i < (size_t)WELEMS; i += (size_t)256 * 1024)
        s += fabsf(w[i]);
    sm[tid] = s;
    __syncthreads();
    for (int off = 128; off > 0; off >>= 1) {
        if (tid < off) sm[tid] += sm[tid + off];
        __syncthreads();
    }
    if (tid == 0) partials[blockIdx.x] = sm[0];
}

__global__ __launch_bounds__(256)
void bitnet_alpha_final(const float* __restrict__ partials, float* __restrict__ alphap) {
    __shared__ float sm[256];
    const int tid = threadIdx.x;
    sm[tid] = partials[tid] + partials[tid + 256] + partials[tid + 512] + partials[tid + 768];
    __syncthreads();
    for (int off = 128; off > 0; off >>= 1) {
        if (tid < off) sm[tid] += sm[tid + off];
        __syncthreads();
    }
    if (tid == 0) *alphap = sm[0] / (float)WELEMS + 1e-8f;
}

// ---------------- 2) quantize W to ternary f16 (unscaled, exact) -----------

__global__ __launch_bounds__(256)
void bitnet_quant_w(const float* __restrict__ w, const float* __restrict__ alphap,
                    _Float16* __restrict__ wh) {
    const float inva = 1.0f / (*alphap);
    const size_t i = ((size_t)blockIdx.x * 256 + threadIdx.x) * 4;
    v4f v = *(const v4f*)(w + i);
    v4h o;
#pragma unroll
    for (int j = 0; j < 4; ++j) {
        float r = __builtin_rintf(v[j] * inva);      // round-to-nearest-even
        r = fminf(1.f, fmaxf(-1.f, r));              // ternary {-1,0,1}
        o[j] = (_Float16)r;                          // exact in f16
    }
    *(v4h*)(wh + i) = o;
}

// ---------------- 3) convert X to f16 --------------------------------------

__global__ __launch_bounds__(256)
void bitnet_conv_x(const float* __restrict__ x, _Float16* __restrict__ xh) {
    const size_t i = ((size_t)blockIdx.x * 256 + threadIdx.x) * 4;
    v4f v = *(const v4f*)(x + i);
    v4h o;
#pragma unroll
    for (int j = 0; j < 4; ++j) o[j] = (_Float16)v[j];
    *(v4h*)(xh + i) = o;
}

// ---------------- 4) WMMA GEMM: out = alpha * (Xh @ Wh^T) ------------------
// Block = 256 threads = 8 waves (wave32). Wave tile 64x64, block tile 256x128.
// A (X):  row-major [NTOK, IN_F]  -> lane m=lane%16, K halves per ISA layout
// B (Wh): row-major [OUT_F, IN_F] -> column n of B^T == row n of Wh (K-contig)

__global__ __launch_bounds__(256)
void bitnet_wmma_gemm(const _Float16* __restrict__ X,
                      const _Float16* __restrict__ Wq,
                      const float* __restrict__ alphap,
                      float* __restrict__ out) {
    const int lane  = threadIdx.x & 31;
    const int wid   = threadIdx.x >> 5;
    const int g     = lane >> 4;       // half-wave group (0/1)
    const int ln    = lane & 15;
    const int waveM = wid & 3;         // 4 waves along M
    const int waveN = wid >> 2;        // 2 waves along N
    const int m0 = blockIdx.y * 256 + waveM * 64;
    const int n0 = blockIdx.x * 128 + waveN * 64;

    v8f acc[4][4] = {};

    const _Float16* aBase[4];
    const _Float16* bBase[4];
#pragma unroll
    for (int mi = 0; mi < 4; ++mi)
        aBase[mi] = X + (size_t)(m0 + mi * 16 + ln) * IN_F;
#pragma unroll
    for (int ni = 0; ni < 4; ++ni)
        bBase[ni] = Wq + (size_t)(n0 + ni * 16 + ln) * IN_F;

    for (int k0 = 0; k0 < IN_F; k0 += 32) {
        // Unconditional speculative prefetch of the next K tile: keeps the
        // inner loop branch-free; OOB prefetch past the buffer end is
        // silently dropped (speculative global_prefetch_b8 per CDNA5 ISA).
        __builtin_prefetch(aBase[0] + k0 + 32, 0, 3);
        __builtin_prefetch(bBase[0] + k0 + 32, 0, 3);

        v16h a[4], b[4];
#pragma unroll
        for (int mi = 0; mi < 4; ++mi) {
            // g==0: K {0..7}|{16..23}; g==1: K {8..15}|{24..31}  (ISA 16-bit A layout)
            const _Float16* p = aBase[mi] + k0 + g * 8;
            v8h lo = *(const v8h*)p;
            v8h hi = *(const v8h*)(p + 16);
            a[mi] = __builtin_shufflevector(lo, hi, 0, 1, 2, 3, 4, 5, 6, 7,
                                                    8, 9, 10, 11, 12, 13, 14, 15);
        }
#pragma unroll
        for (int ni = 0; ni < 4; ++ni) {
            // g==0: K {0..7}|{16..23}; g==1: K {8..15}|{24..31}  (matches A)
            const _Float16* q = bBase[ni] + k0 + g * 8;
            v8h blo = *(const v8h*)q;
            v8h bhi = *(const v8h*)(q + 16);
            b[ni] = __builtin_shufflevector(blo, bhi, 0, 1, 2, 3, 4, 5, 6, 7,
                                                      8, 9, 10, 11, 12, 13, 14, 15);
        }
#pragma unroll
        for (int mi = 0; mi < 4; ++mi)
#pragma unroll
            for (int ni = 0; ni < 4; ++ni)
                acc[mi][ni] = __builtin_amdgcn_wmma_f32_16x16x32_f16(
                    false, a[mi], false, b[ni], (short)0, acc[mi][ni], false, false);
    }

    const float alpha = *alphap;
#pragma unroll
    for (int mi = 0; mi < 4; ++mi) {
#pragma unroll
        for (int ni = 0; ni < 4; ++ni) {
            const int n = n0 + ni * 16 + ln;
#pragma unroll
            for (int r = 0; r < 8; ++r) {
                const int m = m0 + mi * 16 + r + 8 * g;   // ISA f32 C/D layout
                out[(size_t)m * OUT_F + n] = alpha * acc[mi][ni][r];
            }
        }
    }
}

// ---------------------------------------------------------------------------

extern "C" void kernel_launch(void* const* d_in, const int* in_sizes, int n_in,
                              void* d_out, int out_size, void* d_ws, size_t ws_size,
                              hipStream_t stream) {
    const float* x = (const float*)d_in[0];   // [NTOK, IN_F]
    const float* w = (const float*)d_in[1];   // [OUT_F, IN_F]
    float* out = (float*)d_out;               // [NTOK, OUT_F]

    char* ws = (char*)d_ws;
    float*    partials = (float*)ws;                                   // 4 KB
    float*    alphap   = (float*)(ws + 4096);                          // 4 B
    _Float16* xh       = (_Float16*)(ws + 8192);                       // 32 MB
    _Float16* wh       = (_Float16*)(ws + 8192 + (size_t)XELEMS * 2);  // 16 MB

    bitnet_abs_partial<<<1024, 256, 0, stream>>>(w, partials);
    bitnet_alpha_final<<<1, 256, 0, stream>>>(partials, alphap);
    bitnet_quant_w<<<WELEMS / 1024, 256, 0, stream>>>(w, alphap, wh);
    bitnet_conv_x<<<XELEMS / 1024, 256, 0, stream>>>(x, xh);

    dim3 grid(OUT_F / 128, NTOK / 256);  // (32, 32)
    bitnet_wmma_gemm<<<grid, 256, 0, stream>>>(xh, wh, alphap, out);
}
